// DIEN_51539607699
// MI455X (gfx1250) — compile-verified, matrix operand order checked by reading, not run on hardware
//
#include <hip/hip_runtime.h>
#include <hip/hip_bf16.h>
#include <stdint.h>

#define B_TOT 256
#define T_SEQ 1024
#define E_DIM 64
#define U_DIM 64
#define U3 192
#define U2 128
#define BT 16                     // batch tile rows per workgroup
#define NBLOCKS (B_TOT / BT)      // 16
#define NTHREADS 384              // 12 waves (wave32)
#define ROW_LD 72                 // bf16 row stride (144B, 16B-aligned)

typedef __attribute__((ext_vector_type(16))) __bf16 v16bf;
typedef __attribute__((ext_vector_type(8)))  float  v8f;
typedef __attribute__((ext_vector_type(16))) unsigned short v16us;
typedef __attribute__((ext_vector_type(8)))  unsigned short v8us;

__device__ __forceinline__ unsigned short f2bf(float f) {
  unsigned int u = __float_as_uint(f);
  unsigned int r = u + 0x7FFFu + ((u >> 16) & 1u);   // round-to-nearest-even
  return (unsigned short)(r >> 16);
}
__device__ __forceinline__ float bf2f(unsigned short s) {
  return __uint_as_float(((unsigned int)s) << 16);
}
__device__ __forceinline__ float sigm(float x) {
  return __builtin_amdgcn_rcpf(1.0f + __expf(-x));
}
__device__ __forceinline__ int kpat(int r, int half) {
  return ((r & 4) << 2) + (half << 3) + ((r & 3) << 1);
}

// A fragment (16x32 bf16 tile) from bf16 rows in LDS: two contiguous b128 loads.
// CDNA5 A layout: lane half 0 -> K {0..7,16..23}, half 1 -> K {8..15,24..31}.
__device__ __forceinline__ v16bf fragA(const unsigned short* src, int kOff, int lane) {
  int m = lane & 15, half = lane >> 4;
  const unsigned short* p = src + m * ROW_LD + kOff + half * 8;
  v8us lo = *(const v8us*)(p);
  v8us hi = *(const v8us*)(p + 16);
  v16us u = __builtin_shufflevector(lo, hi, 0, 1, 2, 3, 4, 5, 6, 7,
                                    8, 9, 10, 11, 12, 13, 14, 15);
  return __builtin_bit_cast(v16bf, u);
}

// B fragment (Kx16 column tile of row-major KxN f32 weight in global) — one-time.
__device__ __forceinline__ v16bf fragB_g(const float* W, int ldw, int kOff, int n0, int lane) {
  int n = n0 + (lane & 15), half = lane >> 4;
  v16us u;
#pragma unroll
  for (int r = 0; r < 8; ++r) {
    int k = kOff + kpat(r, half);
    u[2 * r]     = f2bf(W[(size_t)k * ldw + n]);
    u[2 * r + 1] = f2bf(W[(size_t)(k + 1) * ldw + n]);
  }
  return __builtin_bit_cast(v16bf, u);
}

__device__ __forceinline__ v8f wmma_bf16(v16bf a, v16bf b, v8f c) {
  return __builtin_amdgcn_wmma_f32_16x16x32_bf16(false, a, false, b, (short)0, c, false, false);
}

__device__ __forceinline__ unsigned int lds_off(const void* p) {
  return (unsigned int)(unsigned long long)p;   // low 32 bits of flat shared addr = LDS offset
}

struct LdsPool {
  unsigned short hbf[2][BT][ROW_LD];   // bf16 hidden state, double buffered
  unsigned short kbuf[2][BT][ROW_LD];  // bf16 keys tile, double buffered
  unsigned short obuf[2][BT][ROW_LD];  // bf16 outs tile, double buffered
  float xacc[BT][U3];
  float racc[BT][U3];
  float qw[BT][U_DIM];
  float avec[BT];
  float red[BT][18];
};

__global__ __launch_bounds__(NTHREADS) void dien_fused(
    const float* __restrict__ queries, const float* __restrict__ keys,
    const float* __restrict__ W_att,  const float* __restrict__ gWx,
    const float* __restrict__ gWh,    const float* __restrict__ gbin,
    const float* __restrict__ gbrec,  const float* __restrict__ aWx,
    const float* __restrict__ aWh,    const float* __restrict__ ab,
    float* __restrict__ out, unsigned short* __restrict__ outs_ws,
    float* __restrict__ lg_ws, unsigned short* __restrict__ keysbf)
{
  __shared__ LdsPool lds;
  const int tid  = threadIdx.x;
  const int lane = tid & 31;
  const int wv   = tid >> 5;           // 0..11
  const int b0   = blockIdx.x * BT;
  const int n0   = wv * 16;            // this wave's N-tile origin

  // -------- prepass: convert this block's keys slice to bf16 in workspace --------
  {
    const float* src = keys + (size_t)b0 * T_SEQ * E_DIM;      // 16 contiguous rows
    unsigned short* dst = keysbf + (size_t)b0 * T_SEQ * E_DIM;
    const int total4 = BT * T_SEQ * E_DIM / 4;
    for (int i = tid; i < total4; i += NTHREADS) {
      float4 v = ((const float4*)src)[i];
      union { unsigned short us[4]; uint2 u2; } pk;
      pk.us[0] = f2bf(v.x); pk.us[1] = f2bf(v.y);
      pk.us[2] = f2bf(v.z); pk.us[3] = f2bf(v.w);
      ((uint2*)dst)[i] = pk.u2;
    }
  }

  // -------- init: qw = queries @ W_att^T, zero hbf (both buffers incl. padding) ---
  for (int i = tid; i < BT * U_DIM; i += NTHREADS) {
    int b = i >> 6, u = i & 63;
    const float* q = queries + (size_t)(b0 + b) * U_DIM;
    const float* w = W_att + (size_t)u * U_DIM;
    float s = 0.f;
#pragma unroll 8
    for (int k = 0; k < U_DIM; ++k) s += q[k] * w[k];
    lds.qw[b][u] = s;
  }
  for (int i = tid; i < 2 * BT * ROW_LD; i += NTHREADS) ((unsigned short*)lds.hbf)[i] = 0;

  // -------- per-thread gate element slots (fixed (b,u) per slot, fully scalar) ----
  const int i0 = tid, i1 = tid + NTHREADS, i2 = tid + 2 * NTHREADS;
  const bool has2 = (i2 < BT * U_DIM);
  const int e0b = i0 >> 6, e0u = i0 & 63;
  const int e1b = i1 >> 6, e1u = i1 & 63;
  const int e2b = (has2 ? (i2 >> 6) : 0), e2u = i2 & 63;
  float h0 = 0.f, h1 = 0.f, h2 = 0.f;
  const float bz0 = gbin[e0u] + gbrec[e0u], bz1 = gbin[e1u] + gbrec[e1u], bz2 = gbin[e2u] + gbrec[e2u];
  const float br0 = gbin[64 + e0u] + gbrec[64 + e0u], br1 = gbin[64 + e1u] + gbrec[64 + e1u],
              br2 = gbin[64 + e2u] + gbrec[64 + e2u];
  const float bxh0 = gbin[128 + e0u], bxh1 = gbin[128 + e1u], bxh2 = gbin[128 + e2u];
  const float brh0 = gbrec[128 + e0u], brh1 = gbrec[128 + e1u], brh2 = gbrec[128 + e2u];
  const float abr0 = ab[e0u], abr1 = ab[e1u], abr2 = ab[e2u];
  const float abc0 = ab[64 + e0u], abc1 = ab[64 + e1u], abc2 = ab[64 + e2u];

  auto gru_gate = [&](int b, int u, float h, float bzv, float brv, float bxhv,
                      float brhv, int cur) -> float {
    float z  = sigm(lds.xacc[b][u]      + lds.racc[b][u]      + bzv);
    float r  = sigm(lds.xacc[b][64 + u] + lds.racc[b][64 + u] + brv);
    float hh = fmaxf(lds.xacc[b][128 + u] + bxhv + r * (lds.racc[b][128 + u] + brhv), 0.f);
    float hn = z * h + (1.f - z) * hh;
    lds.hbf[cur][b][u] = f2bf(hn);
    return hn;
  };
  auto agru_gate = [&](int b, int u, float h, float brv, float bcv, int cur) -> float {
    float r = sigm(lds.xacc[b][u] + lds.racc[b][u] + brv);
    float c = fmaxf(lds.xacc[b][64 + u] + bcv + r * lds.racc[b][64 + u], 0.f);
    float a = lds.avec[b];
    float hn = (1.f - a) * h + a * c;
    lds.hbf[cur][b][u] = f2bf(hn);
    return hn;
  };

  // -------- GRU weight B-fragments (each wave: one 16-col N tile of 192) ---------
  v16bf wx0 = fragB_g(gWx, U3, 0,  n0, lane);
  v16bf wx1 = fragB_g(gWx, U3, 32, n0, lane);
  v16bf wh0 = fragB_g(gWh, U3, 0,  n0, lane);
  v16bf wh1 = fragB_g(gWh, U3, 32, n0, lane);

  // async copy of keysbf[b0.., t, :] (bf16, 2KB) into kbuf[buf]; waves 0..3 issue
  auto issue_keys = [&](int t, int buf) {
    if (wv < 4) {
      int unit = wv * 32 + lane;       // 0..127
      int row = unit >> 3, chunk = unit & 7;
      unsigned int loff = lds_off(&lds.kbuf[buf][row][chunk * 8]);
      const void* g = (const void*)(keysbf + ((size_t)(b0 + row) * T_SEQ + t) * E_DIM + chunk * 8);
      asm volatile("global_load_async_to_lds_b128 %0, %1, off" :: "v"(loff), "v"(g) : "memory");
    }
  };
  // async copy of outs_ws[t, b0.., :] (bf16, 2KB) into obuf[buf]; waves 0..3 issue
  auto issue_outs = [&](int t, int buf) {
    if (wv < 4) {
      int unit = wv * 32 + lane;
      int row = unit >> 3, chunk = unit & 7;
      unsigned int loff = lds_off(&lds.obuf[buf][row][chunk * 8]);
      const void* g = (const void*)(outs_ws + ((size_t)t * B_TOT + b0 + row) * U_DIM + chunk * 8);
      asm volatile("global_load_async_to_lds_b128 %0, %1, off" :: "v"(loff), "v"(g) : "memory");
    }
  };
  // async store of hbf[buf] -> outs_ws[t]; waves 8..11 issue
  auto store_outs = [&](int t, int buf) {
    if (wv >= 8) {
      int unit = (wv - 8) * 32 + lane;
      int row = unit >> 3, chunk = unit & 7;
      unsigned int loff = lds_off(&lds.hbf[buf][row][chunk * 8]);
      void* g = (void*)(outs_ws + ((size_t)t * B_TOT + b0 + row) * U_DIM + chunk * 8);
      asm volatile("global_store_async_from_lds_b128 %0, %1, off" :: "v"(g), "v"(loff) : "memory");
    }
  };

  __threadfence();          // keysbf visible to async engine
  issue_keys(0, 0);
  __syncthreads();

  // ================= Phase A: GRU scan over T, logits on the fly =================
  for (int t = 0; t < T_SEQ; ++t) {
    const int cur = t & 1, prv = cur ^ 1;
    issue_keys(t + 1 < T_SEQ ? t + 1 : t, prv);
    // one wait serves all roles: loaders (kbuf[cur] ready), storers (hbf[cur] free)
    asm volatile("s_wait_asynccnt 1" ::: "memory");
    __syncthreads();

    v16bf ak0 = fragA(&lds.kbuf[cur][0][0], 0,  lane);
    v16bf ak1 = fragA(&lds.kbuf[cur][0][0], 32, lane);
    v16bf ah0 = fragA(&lds.hbf[prv][0][0],  0,  lane);
    v16bf ah1 = fragA(&lds.hbf[prv][0][0],  32, lane);
    v8f ax = {}; ax = wmma_bf16(ak0, wx0, ax); ax = wmma_bf16(ak1, wx1, ax);
    v8f rr = {}; rr = wmma_bf16(ah0, wh0, rr); rr = wmma_bf16(ah1, wh1, rr);
    {
      int n = lane & 15, half = lane >> 4;
#pragma unroll
      for (int r = 0; r < 8; ++r) {
        int m = r + half * 8;
        lds.xacc[m][n0 + n] = ax[r];
        lds.racc[m][n0 + n] = rr[r];
      }
    }
    __syncthreads();

    // gates: f32 state in fixed registers, bf16 mirror in LDS
    h0 = gru_gate(e0b, e0u, h0, bz0, br0, bxh0, brh0, cur);
    h1 = gru_gate(e1b, e1u, h1, bz1, br1, bxh1, brh1, cur);
    if (has2) h2 = gru_gate(e2b, e2u, h2, bz2, br2, bxh2, brh2, cur);
    __syncthreads();

    store_outs(t, cur);                        // async LDS -> global (waves 8..11)
    if (tid >= 128 && tid < 128 + BT) {        // logit[b][t] = h . qw
      int b = tid - 128;
      float s = 0.f;
#pragma unroll 8
      for (int u = 0; u < U_DIM; ++u) s += bf2f(lds.hbf[cur][b][u]) * lds.qw[b][u];
      lg_ws[(size_t)(b0 + b) * T_SEQ + t] = s;
    }
  }

  asm volatile("s_wait_asynccnt 0" ::: "memory");
  __threadfence();
  __syncthreads();

  // ================= Phase B: softmax over T per batch row (in lg_ws) ============
  if (tid < 256) {
    int b = tid >> 4, sub = tid & 15;
    const float* row = lg_ws + (size_t)(b0 + b) * T_SEQ;
    float mx = -3.4e38f;
    for (int t = sub; t < T_SEQ; t += 16) mx = fmaxf(mx, row[t]);
    lds.red[b][sub] = mx;
  }
  __syncthreads();
  if (tid < BT) {
    float mx = lds.red[tid][0];
    for (int j = 1; j < 16; ++j) mx = fmaxf(mx, lds.red[tid][j]);
    lds.red[tid][16] = mx;
  }
  __syncthreads();
  if (tid < 256) {
    int b = tid >> 4, sub = tid & 15;
    const float* row = lg_ws + (size_t)(b0 + b) * T_SEQ;
    float mx = lds.red[b][16], s = 0.f;
    for (int t = sub; t < T_SEQ; t += 16) s += __expf(row[t] - mx);
    lds.red[b][sub] = s;
  }
  __syncthreads();
  if (tid < BT) {
    float s = 0.f;
    for (int j = 0; j < 16; ++j) s += lds.red[tid][j];
    lds.red[tid][17] = s;
  }
  __syncthreads();
  if (tid < 256) {
    int b = tid >> 4, sub = tid & 15;
    float* row = lg_ws + (size_t)(b0 + b) * T_SEQ;
    float mx = lds.red[b][16], inv = __builtin_amdgcn_rcpf(lds.red[b][17]);
    for (int t = sub; t < T_SEQ; t += 16) row[t] = __expf(row[t] - mx) * inv;
  }
  __threadfence();

  // ================= Phase C: AUGRU scan =================
  v16bf awx0 = {}, awx1 = {}, awh0 = {}, awh1 = {};
  if (wv < 8) {                          // 8 N-tiles cover 2U = 128 columns
    awx0 = fragB_g(aWx, U2, 0,  n0, lane);
    awx1 = fragB_g(aWx, U2, 32, n0, lane);
    awh0 = fragB_g(aWh, U2, 0,  n0, lane);
    awh1 = fragB_g(aWh, U2, 32, n0, lane);
  }
  h0 = h1 = h2 = 0.f;
  for (int i = tid; i < 2 * BT * ROW_LD; i += NTHREADS) ((unsigned short*)lds.hbf)[i] = 0;
  issue_outs(0, 0);
  __syncthreads();

  for (int t = 0; t < T_SEQ; ++t) {
    const int cur = t & 1, prv = cur ^ 1;
    issue_outs(t + 1 < T_SEQ ? t + 1 : t, prv);
    asm volatile("s_wait_asynccnt 1" ::: "memory");
    if (tid < BT) lds.avec[tid] = lg_ws[(size_t)(b0 + tid) * T_SEQ + t];
    __syncthreads();

    if (wv < 8) {
      v16bf ao0 = fragA(&lds.obuf[cur][0][0], 0,  lane);
      v16bf ao1 = fragA(&lds.obuf[cur][0][0], 32, lane);
      v16bf ah0 = fragA(&lds.hbf[prv][0][0],  0,  lane);
      v16bf ah1 = fragA(&lds.hbf[prv][0][0],  32, lane);
      v8f ax = {}; ax = wmma_bf16(ao0, awx0, ax); ax = wmma_bf16(ao1, awx1, ax);
      v8f rr = {}; rr = wmma_bf16(ah0, awh0, rr); rr = wmma_bf16(ah1, awh1, rr);
      int n = lane & 15, half = lane >> 4;
#pragma unroll
      for (int r = 0; r < 8; ++r) {
        int m = r + half * 8;
        lds.xacc[m][n0 + n] = ax[r];
        lds.racc[m][n0 + n] = rr[r];
      }
    }
    __syncthreads();

    h0 = agru_gate(e0b, e0u, h0, abr0, abc0, cur);
    h1 = agru_gate(e1b, e1u, h1, abr1, abc1, cur);
    if (has2) h2 = agru_gate(e2b, e2u, h2, abr2, abc2, cur);
    __syncthreads();
  }

  // final hidden state (exact f32 from registers)
  out[(size_t)(b0 + e0b) * U_DIM + e0u] = h0;
  out[(size_t)(b0 + e1b) * U_DIM + e1u] = h1;
  if (has2) out[(size_t)(b0 + e2b) * U_DIM + e2u] = h2;
}

extern "C" void kernel_launch(void* const* d_in, const int* in_sizes, int n_in,
                              void* d_out, int out_size, void* d_ws, size_t ws_size,
                              hipStream_t stream) {
  (void)in_sizes; (void)n_in; (void)out_size; (void)ws_size;
  const float* queries = (const float*)d_in[0];
  const float* keys    = (const float*)d_in[1];
  const float* W_att   = (const float*)d_in[2];
  const float* gWx     = (const float*)d_in[3];
  const float* gWh     = (const float*)d_in[4];
  const float* gbin    = (const float*)d_in[5];
  const float* gbrec   = (const float*)d_in[6];
  const float* aWx     = (const float*)d_in[7];
  const float* aWh     = (const float*)d_in[8];
  const float* ab      = (const float*)d_in[9];
  const size_t outs_bytes = (size_t)T_SEQ * B_TOT * U_DIM * 2;   // 32 MB bf16 outs
  const size_t lg_bytes   = (size_t)B_TOT * T_SEQ * 4;           // 1 MB logits/attn
  unsigned short* outs_ws = (unsigned short*)d_ws;
  float* lg_ws = (float*)((char*)d_ws + outs_bytes);
  unsigned short* keysbf = (unsigned short*)((char*)d_ws + outs_bytes + lg_bytes);
  dien_fused<<<NBLOCKS, NTHREADS, 0, stream>>>(queries, keys, W_att, gWx, gWh,
      gbin, gbrec, aWx, aWh, ab, (float*)d_out, outs_ws, lg_ws, keysbf);
}
